// SelfAttention_9706626089927
// MI455X (gfx1250) — matrix operand order repaired and checked by reading, not compiled, not run on hardware
//
#include <hip/hip_runtime.h>

// ---------------------------------------------------------------------------
// Problem constants (from reference): b=2, L=2048, d=1024, 16 heads, 64 ch.
// ---------------------------------------------------------------------------
#define BATCH 2
#define SEQ   2048
#define DIM   1024
#define NH    16
#define CH    64
#define MTOT  (BATCH * SEQ)           // 4096 rows
#define QK_SCALE 0.35355339059327379f // 64^(-0.25)

typedef __attribute__((ext_vector_type(16))) __bf16 v16bf;
typedef __attribute__((ext_vector_type(8)))  __bf16 v8bf;
typedef __attribute__((ext_vector_type(8)))  float  v8f;
typedef __attribute__((ext_vector_type(4)))  int    v4i;

#define AS1 __attribute__((address_space(1)))
#define AS3 __attribute__((address_space(3)))

#if __has_builtin(__builtin_amdgcn_global_load_async_to_lds_b128)
#define HAS_ASYNC_LDS 1
#else
#define HAS_ASYNC_LDS 0
#endif

// ---------------------------------------------------------------------------
// Helpers
// ---------------------------------------------------------------------------
__device__ __forceinline__ unsigned short f32_to_bf16(float f) {
  union { float f; unsigned u; } v; v.f = f;
  unsigned u = v.u;
  u += 0x7fffu + ((u >> 16) & 1u);       // round-to-nearest-even
  return (unsigned short)(u >> 16);
}

__device__ __forceinline__ v8f v8f_splat(float x) {
  v8f v;
#pragma unroll
  for (int i = 0; i < 8; ++i) v[i] = x;
  return v;
}

// 16-byte global -> LDS copy; async (ASYNCcnt-tracked) when available.
__device__ __forceinline__ void stage16(const unsigned short* g,
                                        unsigned short* l) {
#if HAS_ASYNC_LDS
  __builtin_amdgcn_global_load_async_to_lds_b128(
      (AS1 v4i*)g, (AS3 v4i*)l, 0, 0);
#else
  *(uint4*)l = *(const uint4*)g;
#endif
}

__device__ __forceinline__ void wait_stage() {
#if HAS_ASYNC_LDS
  asm volatile("s_wait_asynccnt 0x0" ::: "memory");
#endif
  __syncthreads();
}

// A-fragment (16x32 bf16, row-major source, leading dim `ld` elements).
// Layout: lane m = t&15, half = t>>4; element s -> K = half*16 + s
// (contiguous order, matching the B fragment) => two 8-element chunks.
__device__ __forceinline__ v16bf load_a_frag(const unsigned short* base,
                                             int ld, int k0) {
  int lane = threadIdx.x & 31;
  int m = lane & 15, half = lane >> 4;
  const v8bf* c0 = (const v8bf*)(base + (size_t)m * ld + k0 + half * 16);
  const v8bf* c1 = (const v8bf*)(base + (size_t)m * ld + k0 + 8 + half * 16);
  v8bf x0 = *c0, x1 = *c1;
  v16bf r;
#pragma unroll
  for (int i = 0; i < 8; ++i) { r[i] = x0[i]; r[i + 8] = x1[i]; }
  return r;
}

// B-fragment (32x16 bf16) from an N-major source: element [n][k] lives at
// base + n*ld + k. Lane n = t&15, K-base = (t>>4)*16 -> one 32-byte load.
__device__ __forceinline__ v16bf load_b_frag_nmajor(const unsigned short* base,
                                                    int ld, int k0) {
  int lane = threadIdx.x & 31;
  int n = lane & 15, kb = (lane >> 4) * 16;
  return *(const v16bf*)(base + (size_t)n * ld + k0 + kb);
}

__device__ __forceinline__ v8f wmma_bf16(v16bf a, v16bf b, v8f c) {
  return __builtin_amdgcn_wmma_f32_16x16x32_bf16(
      false, a, false, b, (short)0, c, false, false);
}

// ---------------------------------------------------------------------------
// fp32 -> bf16 conversion
// ---------------------------------------------------------------------------
__global__ void cvt_f32_bf16(const float* __restrict__ in,
                             unsigned short* __restrict__ out, int n) {
  int i = blockIdx.x * blockDim.x + threadIdx.x;
  int stride = gridDim.x * blockDim.x;
  for (; i < n; i += stride) out[i] = f32_to_bf16(in[i]);
}

// ---------------------------------------------------------------------------
// GEMM 1: qkv = x @ w_qkv^T + b_qkv    (M=4096, N=3072, K=1024)
// One wave -> 16x64 output tile. Epilogue scatters into Q, K (scaled) and
// V-transposed bf16 buffers for the attention kernel.
// ---------------------------------------------------------------------------
__global__ void gemm_qkv(const unsigned short* __restrict__ xb,
                         const unsigned short* __restrict__ wqb,
                         const float* __restrict__ bias,
                         unsigned short* __restrict__ Qb,
                         unsigned short* __restrict__ Kb,
                         unsigned short* __restrict__ Vt) {
  int wave = blockIdx.x * (blockDim.x >> 5) + (threadIdx.x >> 5);
  int mt = wave & 255;          // 256 M-tiles
  int ns = wave >> 8;           // 48 N-strips of 64
  int lane = threadIdx.x & 31;
  int nl = lane & 15, half = lane >> 4;

  v8f acc[4];
#pragma unroll
  for (int f = 0; f < 4; ++f)
    acc[f] = v8f_splat(bias[ns * 64 + f * 16 + nl]);   // C = bias per column

  const unsigned short* arow = xb + (size_t)mt * 16 * DIM;
  for (int k0 = 0; k0 < DIM; k0 += 32) {
    __builtin_prefetch(arow + (size_t)(lane & 15) * DIM + k0 + 64, 0, 3);
    v16bf a = load_a_frag(arow, DIM, k0);
#pragma unroll
    for (int f = 0; f < 4; ++f) {
      v16bf b = load_b_frag_nmajor(wqb + (size_t)(ns * 64 + f * 16) * DIM,
                                   DIM, k0);
      acc[f] = wmma_bf16(a, b, acc[f]);
    }
  }

  int mbase = mt * 16;
#pragma unroll
  for (int f = 0; f < 4; ++f) {
    int ng = ns * 64 + f * 16 + nl;        // 0..3071
    int which = ng >> 10;                  // 0=q 1=k 2=v
    int f1 = ng & 1023;
    int h = f1 >> 6, c = f1 & 63;
#pragma unroll
    for (int r = 0; r < 8; ++r) {
      int mg = mbase + r + 8 * half;       // global token row
      int bi = mg >> 11, li = mg & 2047;
      float v = acc[f][r];
      if (which == 0)
        Qb[(((size_t)(bi * NH + h)) * SEQ + li) * CH + c] =
            f32_to_bf16(v * QK_SCALE);
      else if (which == 1)
        Kb[(((size_t)(bi * NH + h)) * SEQ + li) * CH + c] =
            f32_to_bf16(v * QK_SCALE);
      else
        Vt[(((size_t)(bi * NH + h)) * CH + c) * SEQ + li] = f32_to_bf16(v);
    }
  }
}

// ---------------------------------------------------------------------------
// Flash attention, block-cooperative: one block = one (batch, head) and a
// 128-row query panel (8 waves x 16 rows). Each 32-key K/V tile is staged
// into LDS ONCE per block via async global->LDS copies (double-buffered,
// overlapped with WMMA + online-softmax compute), then all 8 waves consume
// it as WMMA B-fragments straight from LDS.
// ---------------------------------------------------------------------------
__global__ void attn_fwd(const unsigned short* __restrict__ Qb,
                         const unsigned short* __restrict__ Kb,
                         const unsigned short* __restrict__ Vt,
                         unsigned short* __restrict__ AOb) {
  __shared__ unsigned short Ks[2][32 * CH];   // [key][ch]  2 x 4 KB
  __shared__ unsigned short Vs[2][CH * 32];   // [ch][key]  2 x 4 KB
  __shared__ unsigned short Pt[8][16 * 32];   // P bounce,  8 x 1 KB

  int wavein = threadIdx.x >> 5;
  int qb = blockIdx.x & 15;       // 16 query panels of 128 rows
  int bh = blockIdx.x >> 4;       // 32 (batch, head) pairs
  int bi = bh >> 4, h = bh & 15;
  int q0 = qb * 128 + wavein * 16;
  int lane = threadIdx.x & 31;
  int nl = lane & 15, half = lane >> 4;
  int t = threadIdx.x;

  const unsigned short* Qp = Qb + (((size_t)(bi * NH + h)) * SEQ + q0) * CH;
  const unsigned short* Kh = Kb + ((size_t)(bi * NH + h)) * SEQ * CH;
  const unsigned short* Vh = Vt + ((size_t)(bi * NH + h)) * CH * SEQ;

  v16bf aq0 = load_a_frag(Qp, CH, 0);
  v16bf aq1 = load_a_frag(Qp, CH, 32);

  float rm[8], rl[8];
  v8f o[4];
#pragma unroll
  for (int r = 0; r < 8; ++r) { rm[r] = -3.0e38f; rl[r] = 0.0f; }
#pragma unroll
  for (int f = 0; f < 4; ++f) o[f] = v8f_splat(0.0f);

  // Cooperative tile staging: K tile is a contiguous 4 KB slab; V tile is
  // 64 rows x 64 B from the channel-major V. 256 threads x 16 B each.
  auto stage = [&](int buf, int j0) {
    stage16(Kh + (size_t)j0 * CH + t * 8, &Ks[buf][t * 8]);
    stage16(Vh + (size_t)(t >> 2) * SEQ + j0 + (t & 3) * 8,
            &Vs[buf][(t >> 2) * 32 + (t & 3) * 8]);
  };

  int nblk_blkmax = qb * 4 + 4;               // keys needed by last wave
  int nblk_me = (q0 + 16 + 31) >> 5;          // keys needed by this wave

  stage(0, 0);
  wait_stage();

  for (int blk = 0; blk < nblk_blkmax; ++blk) {
    int j0 = blk * 32;
    int cur = blk & 1;
    if (blk + 1 < nblk_blkmax) stage(cur ^ 1, j0 + 32);  // overlap next tile

    if (blk < nblk_me) {                      // wave-uniform causal skip
      const unsigned short* ks = Ks[cur];
      const unsigned short* vs = Vs[cur];

      v8f z = v8f_splat(0.0f);
      v8f s0 = wmma_bf16(aq0, load_b_frag_nmajor(ks, CH, 0), z);
      s0 = wmma_bf16(aq1, load_b_frag_nmajor(ks, CH, 32), s0);
      v8f s1 = wmma_bf16(aq0, load_b_frag_nmajor(ks + 16 * CH, CH, 0), z);
      s1 = wmma_bf16(aq1, load_b_frag_nmajor(ks + 16 * CH, CH, 32), s1);

      if (j0 + 31 > q0) {                     // wave-uniform causal mask
#pragma unroll
        for (int r = 0; r < 8; ++r) {
          int row = q0 + r + 8 * half;
          if (j0 + nl > row)      s0[r] = -1.0e30f;
          if (j0 + 16 + nl > row) s1[r] = -1.0e30f;
        }
      }

      // Online softmax: row reductions across 16 lanes of each half-wave.
      float alpha[8];
#pragma unroll
      for (int r = 0; r < 8; ++r) {
        float bm = fmaxf(s0[r], s1[r]);
        bm = fmaxf(bm, __shfl_xor(bm, 1, 32));
        bm = fmaxf(bm, __shfl_xor(bm, 2, 32));
        bm = fmaxf(bm, __shfl_xor(bm, 4, 32));
        bm = fmaxf(bm, __shfl_xor(bm, 8, 32));
        float mn = fmaxf(rm[r], bm);
        alpha[r] = __expf(rm[r] - mn);
        float p0 = __expf(s0[r] - mn);
        float p1 = __expf(s1[r] - mn);
        s0[r] = p0; s1[r] = p1;
        float ps = p0 + p1;
        ps += __shfl_xor(ps, 1, 32);
        ps += __shfl_xor(ps, 2, 32);
        ps += __shfl_xor(ps, 4, 32);
        ps += __shfl_xor(ps, 8, 32);
        rl[r] = rl[r] * alpha[r] + ps;
        rm[r] = mn;
      }
#pragma unroll
      for (int f = 0; f < 4; ++f)
#pragma unroll
        for (int r = 0; r < 8; ++r) o[f][r] *= alpha[r];

      // P: C-frag layout -> per-wave LDS tile -> A-frag layout.
      unsigned short* pt = Pt[wavein];
#pragma unroll
      for (int r = 0; r < 8; ++r) {
        int m = r + 8 * half;
        pt[m * 32 + nl]      = f32_to_bf16(s0[r]);
        pt[m * 32 + 16 + nl] = f32_to_bf16(s1[r]);
      }
      asm volatile("s_wait_dscnt 0x0" ::: "memory");  // per-wave LDS RAW
      v16bf ap = load_a_frag(pt, 32, 0);

#pragma unroll
      for (int f = 0; f < 4; ++f)
        o[f] = wmma_bf16(ap, load_b_frag_nmajor(vs + f * 16 * 32, 32, 0),
                         o[f]);
    }

    wait_stage();   // next buffer resident + all waves done with `cur`
  }

  // Normalize and store attention output (bf16, [token][d] for GEMM2).
#pragma unroll
  for (int f = 0; f < 4; ++f) {
#pragma unroll
    for (int r = 0; r < 8; ++r) {
      int mg = q0 + r + 8 * half;
      float v = o[f][r] / rl[r];
      AOb[((size_t)bi * SEQ + mg) * DIM + h * CH + f * 16 + nl] =
          f32_to_bf16(v);
    }
  }
}

// ---------------------------------------------------------------------------
// GEMM 2: out = attn_out @ w_proj^T + b_proj   (M=4096, N=1024, K=1024), f32.
// ---------------------------------------------------------------------------
__global__ void gemm_proj(const unsigned short* __restrict__ aob,
                          const unsigned short* __restrict__ wpb,
                          const float* __restrict__ bias,
                          float* __restrict__ out) {
  int wave = blockIdx.x * (blockDim.x >> 5) + (threadIdx.x >> 5);
  int mt = wave & 255;          // 256 M-tiles
  int ns = wave >> 8;           // 16 N-strips of 64
  int lane = threadIdx.x & 31;
  int nl = lane & 15, half = lane >> 4;

  v8f acc[4];
#pragma unroll
  for (int f = 0; f < 4; ++f)
    acc[f] = v8f_splat(bias[ns * 64 + f * 16 + nl]);

  const unsigned short* arow = aob + (size_t)mt * 16 * DIM;
  for (int k0 = 0; k0 < DIM; k0 += 32) {
    __builtin_prefetch(arow + (size_t)(lane & 15) * DIM + k0 + 64, 0, 3);
    v16bf a = load_a_frag(arow, DIM, k0);
#pragma unroll
    for (int f = 0; f < 4; ++f) {
      v16bf b = load_b_frag_nmajor(wpb + (size_t)(ns * 64 + f * 16) * DIM,
                                   DIM, k0);
      acc[f] = wmma_bf16(a, b, acc[f]);
    }
  }

  int mbase = mt * 16;
#pragma unroll
  for (int f = 0; f < 4; ++f) {
    int ng = ns * 64 + f * 16 + nl;
#pragma unroll
    for (int r = 0; r < 8; ++r) {
      int mg = mbase + r + 8 * half;
      out[(size_t)mg * DIM + ng] = acc[f][r];
    }
  }
}

// ---------------------------------------------------------------------------
// Host launch
// ---------------------------------------------------------------------------
extern "C" void kernel_launch(void* const* d_in, const int* in_sizes, int n_in,
                              void* d_out, int out_size, void* d_ws,
                              size_t ws_size, hipStream_t stream) {
  const float* x      = (const float*)d_in[0];
  // d_in[1] = mask (int32): causality is known statically, unused.
  const float* w_qkv  = (const float*)d_in[2];
  const float* b_qkv  = (const float*)d_in[3];
  const float* w_proj = (const float*)d_in[4];
  const float* b_proj = (const float*)d_in[5];
  float* out = (float*)d_out;

  char* ws = (char*)d_ws;
  unsigned short* xb  = (unsigned short*)(ws);                    //  8 MiB
  unsigned short* wqb = (unsigned short*)(ws + (8u << 20));       //  6 MiB
  unsigned short* wpb = (unsigned short*)(ws + (14u << 20));      //  2 MiB
  unsigned short* Qb  = (unsigned short*)(ws + (16u << 20));      //  8 MiB
  unsigned short* Kb  = (unsigned short*)(ws + (24u << 20));      //  8 MiB
  unsigned short* Vt  = (unsigned short*)(ws + (32u << 20));      //  8 MiB
  unsigned short* AOb = (unsigned short*)(ws + (40u << 20));      //  8 MiB

  const int nx = MTOT * DIM;          // 4194304
  const int nwq = 3 * DIM * DIM;      // 3145728
  const int nwp = DIM * DIM;          // 1048576
  cvt_f32_bf16<<<(nx  + 1023) / 1024, 256, 0, stream>>>(x,      xb,  nx);
  cvt_f32_bf16<<<(nwq + 1023) / 1024, 256, 0, stream>>>(w_qkv,  wqb, nwq);
  cvt_f32_bf16<<<(nwp + 1023) / 1024, 256, 0, stream>>>(w_proj, wpb, nwp);

  // 256 M-tiles * 48 N-strips = 12288 waves / 8 per block
  gemm_qkv<<<1536, 256, 0, stream>>>(xb, wqb, b_qkv, Qb, Kb, Vt);

  // 32 (batch,head) * 16 query panels = 512 blocks, 8 waves each
  attn_fwd<<<512, 256, 0, stream>>>(Qb, Kb, Vt, AOb);

  // 256 M-tiles * 16 N-strips = 4096 waves / 8 per block
  gemm_proj<<<512, 256, 0, stream>>>(AOb, wpb, b_proj, out);
}
